// CellMessageLayer_45397804319392
// MI455X (gfx1250) — matrix operand order repaired and checked by reading, not compiled, not run on hardware
//
#include <hip/hip_runtime.h>
#include <hip/hip_bf16.h>

#define HID     256
#define EDGE_F  64
#define N_CELLS 4096
#define N_EDGES 65536
#define BSZ     4

typedef __attribute__((ext_vector_type(16))) _Float16 v16h;
typedef __attribute__((ext_vector_type(8)))  _Float16 v8h;
typedef __attribute__((ext_vector_type(8)))  float    v8f;

// ---------------------------------------------------------------------------
// WMMA fragment helpers (wave32, V_WMMA_F32_16X16X32_F16)
//
// A (16x32 f16): lanes 0-15 = rows 0-15 with K k0..k0+15 contiguous,
//                lanes 16-31 = rows 0-15 with K k0+16..k0+31.
// B (32x16 f16): lane = output column N; lanes 0-15 hold K 0..15 contiguous,
//                lanes 16-31 hold K 16..31.
// D/C (16x16 f32): VGPR v, lanes 0-15 -> (M=v, N=lane); lanes 16-31 ->
//                (M=v+8, N=lane-16).
// ---------------------------------------------------------------------------
__device__ __forceinline__ v16h load_a_frag(const _Float16* base, int r0, int k0, int stride) {
  const int lane = threadIdx.x & 31;
  const _Float16* p = base + (r0 + (lane & 15)) * stride + k0 + ((lane >> 4) << 4);
  union { v16h v; v8h h[2]; } u;
  u.h[0] = *(const v8h*)(p);        // K k..k+7
  u.h[1] = *(const v8h*)(p + 8);    // K k+8..k+15
  return u.v;
}

// B fragment from LDS-staged slice Bs[256][BS_STRIDE] (K-tile of 32)
#define BS_STRIDE 40   // f16 units; 80B row pitch (16B aligned, bank-spread)
__device__ __forceinline__ v16h load_b_lds(const _Float16* Bs, int n0) {
  const int lane = threadIdx.x & 31;
  const _Float16* p = Bs + (n0 + (lane & 15)) * BS_STRIDE + ((lane >> 4) << 4);
  union { v16h v; v8h h[2]; } u;
  u.h[0] = *(const v8h*)(p);        // K 0..7  (or 16..23)
  u.h[1] = *(const v8h*)(p + 8);    // K 8..15 (or 24..31)
  return u.v;
}

// B fragment straight from global [N][K] f16 (node kernel; weights L2-hot)
__device__ __forceinline__ v16h load_b_frag(const _Float16* wT, int n0, int k0, int K) {
  const int lane = threadIdx.x & 31;
  return *(const v16h*)(wT + (size_t)(n0 + (lane & 15)) * K + k0 + ((lane >> 4) << 4));
}

__device__ __forceinline__ v8f wmma_f16(v16h a, v16h b, v8f c) {
  return __builtin_amdgcn_wmma_f32_16x16x32_f16(false, a, false, b, (short)0, c, false, false);
}

// stage one 32-wide K slice of a [N=256][K] f16 weight into Bs (256 threads)
__device__ __forceinline__ void stage_b_slice(const _Float16* __restrict__ wT, int K,
                                              int k0, _Float16* Bs, int t) {
  const _Float16* g = wT + (size_t)t * K + k0;
  v8h x0 = *(const v8h*)(g);
  v8h x1 = *(const v8h*)(g + 8);
  v8h x2 = *(const v8h*)(g + 16);
  v8h x3 = *(const v8h*)(g + 24);
  _Float16* d = Bs + t * BS_STRIDE;
  *(v8h*)(d)      = x0;
  *(v8h*)(d + 8)  = x1;
  *(v8h*)(d + 16) = x2;
  *(v8h*)(d + 24) = x3;
}

// ---------------------------------------------------------------------------
// Prep: transpose + convert fp32 [K][N] weight -> f16 [N][K]
// ---------------------------------------------------------------------------
__global__ void k_transpose_cvt(const float* __restrict__ W, _Float16* __restrict__ wT,
                                int K, int N) {
  int i = blockIdx.x * blockDim.x + threadIdx.x;
  if (i >= K * N) return;
  int k = i / N, n = i - k * N;
  wT[(size_t)n * K + k] = (_Float16)W[i];
}

__global__ void k_zero_f32(float* __restrict__ p, int n) {
  int i = blockIdx.x * blockDim.x + threadIdx.x;
  if (i < n) p[i] = 0.f;
}

__global__ void k_count(const int* __restrict__ eidx, float* __restrict__ cnt) {
  int i = blockIdx.x * blockDim.x + threadIdx.x;
  if (i >= BSZ * N_EDGES) return;
  int b = i >> 16;
  int e = i & (N_EDGES - 1);
  int d = eidx[((size_t)b * N_EDGES + e) * 2 + 1];
  d = d < 0 ? 0 : (d > N_CELLS - 1 ? N_CELLS - 1 : d);
  atomicAdd(&cnt[b * N_CELLS + d], 1.0f);
}

// ---------------------------------------------------------------------------
// Edge message kernel: gather(src) || edge_attr -> MLP(320->256 relu ->256)
// -> atomic scatter-add into agg[b][dst][:].
// Block: 256 threads (8 waves), 128 edges. K-outer GEMM, B staged in LDS
// once per block per K-slice (8x less L2 traffic, 16x A-frag reuse).
// LDS: A[128][320] + H[128][256] + Bs[256][40] (f16) + idx = ~165KB of 320KB.
// ---------------------------------------------------------------------------
__global__ __launch_bounds__(256) void k_edge_msg(
    const float* __restrict__ cell_x, const int* __restrict__ eidx,
    const float* __restrict__ eattr, const _Float16* __restrict__ wT1,
    const float* __restrict__ mb1, const _Float16* __restrict__ wT2,
    const float* __restrict__ mb2, float* __restrict__ agg) {
  extern __shared__ char smem[];
  _Float16* A  = (_Float16*)smem;                            // [128][320]
  _Float16* Hs = (_Float16*)(smem + 128 * 320 * 2);          // [128][256]
  _Float16* Bs = (_Float16*)(smem + 128 * 320 * 2 + 128 * 256 * 2);  // [256][40]
  int* srcL = (int*)(smem + 128 * 320 * 2 + 128 * 256 * 2 + 256 * BS_STRIDE * 2);
  int* dstL = srcL + 128;

  const int t  = threadIdx.x;
  const int b  = blockIdx.y;
  const int e0 = blockIdx.x * 128;

  if (t < 128) {
    long base = ((long)b * N_EDGES + (e0 + t)) * 2;
    int s = eidx[base + 0];
    int d = eidx[base + 1];
    s = s < 0 ? 0 : (s > N_CELLS - 1 ? N_CELLS - 1 : s);
    d = d < 0 ? 0 : (d > N_CELLS - 1 ? N_CELLS - 1 : d);
    srcL[t] = s;
    dstL[t] = d;
  }
  __syncthreads();

  // gather: thread t = column t of cell_x block (256 cols), walk 128 rows
  const size_t cxb = (size_t)b * N_CELLS * HID;
  for (int j = 0; j < 128; ++j)
    A[j * 320 + t] = (_Float16)cell_x[cxb + (size_t)srcL[j] * HID + t];
  const size_t eab = ((size_t)b * N_EDGES + e0) * EDGE_F;
  for (int j = 0; j < 32; ++j) {
    int idx = j * 256 + t;
    int row = idx >> 6, col = idx & 63;
    A[row * 320 + 256 + col] = (_Float16)eattr[eab + (size_t)row * EDGE_F + col];
  }

  const int lane = t & 31;
  const int r0   = (t >> 5) * 16;  // wave's 16-row strip
  const int rb   = r0 + ((lane >> 4) << 3);
  const v8f zero = {};

  // ---- layer 1: [128x320] @ [320x256] + b, relu -> Hs ----
  v8f acc[16];
  #pragma unroll
  for (int i = 0; i < 16; ++i) acc[i] = zero;

  for (int kt = 0; kt < 10; ++kt) {
    __syncthreads();                       // Bs consumed / A-gather done
    stage_b_slice(wT1, 320, kt * 32, Bs, t);
    __syncthreads();
    v16h a = load_a_frag(A, r0, kt * 32, 320);
    #pragma unroll
    for (int nt = 0; nt < 16; ++nt)
      acc[nt] = wmma_f16(a, load_b_lds(Bs, nt * 16), acc[nt]);
  }

  #pragma unroll
  for (int nt = 0; nt < 16; ++nt) {
    const int n = nt * 16 + (lane & 15);
    const float bias = mb1[n];
    #pragma unroll
    for (int v = 0; v < 8; ++v) {
      float x = acc[nt][v] + bias;
      Hs[(rb + v) * HID + n] = (_Float16)(x > 0.f ? x : 0.f);
    }
  }
  __syncthreads();                         // Hs complete

  // ---- layer 2: [128x256] @ [256x256] + b -> atomic scatter ----
  #pragma unroll
  for (int i = 0; i < 16; ++i) acc[i] = zero;

  for (int kt = 0; kt < 8; ++kt) {
    __syncthreads();
    stage_b_slice(wT2, 256, kt * 32, Bs, t);
    __syncthreads();
    v16h a = load_a_frag(Hs, r0, kt * 32, 256);
    #pragma unroll
    for (int nt = 0; nt < 16; ++nt)
      acc[nt] = wmma_f16(a, load_b_lds(Bs, nt * 16), acc[nt]);
  }

  #pragma unroll
  for (int nt = 0; nt < 16; ++nt) {
    const int n = nt * 16 + (lane & 15);
    const float bias = mb2[n];
    #pragma unroll
    for (int v = 0; v < 8; ++v) {
      int row = rb + v;
      atomicAdd(&agg[((size_t)b * N_CELLS + dstL[row]) * HID + n], acc[nt][v] + bias);
    }
  }
}

// ---------------------------------------------------------------------------
// Node update: [cell_x || agg/max(c,1)] -> MLP(512->256 relu ->256)
// + residual + LayerNorm. Block: 128 threads (4 waves), 64 nodes.
// ---------------------------------------------------------------------------
__global__ __launch_bounds__(128) void k_node_update(
    const float* __restrict__ cell_x, const float* __restrict__ agg,
    const float* __restrict__ cnt, const _Float16* __restrict__ uT1,
    const float* __restrict__ ub1, const _Float16* __restrict__ uT2,
    const float* __restrict__ ub2, const float* __restrict__ gamma,
    const float* __restrict__ beta, float* __restrict__ outp) {
  extern __shared__ char smem[];
  _Float16* A  = (_Float16*)smem;                    // [64][512] f16 (phase 1)
  float*    O  = (float*)smem;                       // [64][256] f32 (phase 2, reuses A)
  _Float16* Hs = (_Float16*)(smem + 64 * 512 * 2);   // [64][256]
  float* cntL  = (float*)(smem + 64 * 512 * 2 + 64 * 256 * 2);

  const int t   = threadIdx.x;
  const int b   = blockIdx.y;
  const int n0g = blockIdx.x * 64;

  if (t < 64) {
    float c = cnt[(size_t)b * N_CELLS + n0g + t];
    cntL[t] = c > 1.f ? c : 1.f;
  }
  __syncthreads();

  const size_t cxb = ((size_t)b * N_CELLS + n0g) * HID;
  for (int j = 0; j < 128; ++j) {
    int idx = j * 128 + t;
    int row = idx >> 8, col = idx & 255;
    A[row * 512 + col] = (_Float16)cell_x[cxb + (size_t)row * HID + col];
  }
  for (int j = 0; j < 128; ++j) {
    int idx = j * 128 + t;
    int row = idx >> 8, col = idx & 255;
    A[row * 512 + 256 + col] = (_Float16)(agg[cxb + (size_t)row * HID + col] * (1.f / cntL[row]));
  }
  __syncthreads();

  const int lane = t & 31;
  const int r0   = (t >> 5) * 16;

  // layer 1: [64x512] @ [512x256] + b, relu -> Hs
  for (int nt = 0; nt < 16; ++nt) {
    const int n0 = nt * 16;
    v8f acc = {};
    #pragma unroll
    for (int kt = 0; kt < 16; ++kt)
      acc = wmma_f16(load_a_frag(A, r0, kt * 32, 512),
                     load_b_frag(uT1, n0, kt * 32, 512), acc);
    const int n = n0 + (lane & 15);
    const float bias = ub1[n];
    const int rb = r0 + ((lane >> 4) << 3);
    #pragma unroll
    for (int v = 0; v < 8; ++v) {
      float x = acc[v] + bias;
      Hs[(rb + v) * HID + n] = (_Float16)(x > 0.f ? x : 0.f);
    }
  }
  __syncthreads();

  // layer 2: [64x256] @ [256x256] + b -> O (f32, reuses A region)
  for (int nt = 0; nt < 16; ++nt) {
    const int n0 = nt * 16;
    v8f acc = {};
    #pragma unroll
    for (int kt = 0; kt < 8; ++kt)
      acc = wmma_f16(load_a_frag(Hs, r0, kt * 32, 256),
                     load_b_frag(uT2, n0, kt * 32, 256), acc);
    const int n = n0 + (lane & 15);
    const float bias = ub2[n];
    const int rb = r0 + ((lane >> 4) << 3);
    #pragma unroll
    for (int v = 0; v < 8; ++v)
      O[(rb + v) * HID + n] = acc[v] + bias;
  }
  __syncthreads();

  // residual + LayerNorm: each wave normalizes its own 16 rows, lane-parallel
  for (int rr = 0; rr < 16; ++rr) {
    const int row = r0 + rr;
    const size_t gbase = cxb + (size_t)row * HID;
    const int c0 = lane * 8;
    float h[8];
    float s = 0.f, q = 0.f;
    #pragma unroll
    for (int j = 0; j < 8; ++j) {
      float x = O[row * HID + c0 + j] + cell_x[gbase + c0 + j];
      h[j] = x;
      s += x;
      q += x * x;
    }
    #pragma unroll
    for (int m = 16; m >= 1; m >>= 1) {
      s += __shfl_xor(s, m, 32);
      q += __shfl_xor(q, m, 32);
    }
    const float mu  = s * (1.f / HID);
    const float var = q * (1.f / HID) - mu * mu;
    const float rs  = rsqrtf(var + 1e-5f);
    #pragma unroll
    for (int j = 0; j < 8; ++j)
      outp[gbase + c0 + j] = (h[j] - mu) * rs * gamma[c0 + j] + beta[c0 + j];
  }
}

// ---------------------------------------------------------------------------
// Workspace layout
// ---------------------------------------------------------------------------
static const size_t OFF_WT1 = 0;
static const size_t OFF_WT2 = OFF_WT1 + (size_t)256 * 320 * 2;
static const size_t OFF_UT1 = OFF_WT2 + (size_t)256 * 256 * 2;
static const size_t OFF_UT2 = OFF_UT1 + (size_t)256 * 512 * 2;
static const size_t OFF_AGG = OFF_UT2 + (size_t)256 * 256 * 2;
static const size_t OFF_CNT = OFF_AGG + (size_t)BSZ * N_CELLS * HID * 4;

static const int EDGE_SMEM = 128 * 320 * 2 + 128 * 256 * 2 + 256 * BS_STRIDE * 2
                           + 128 * 2 * 4;                                  // 168960
static const int NODE_SMEM = 64 * 512 * 2 + 64 * 256 * 2 + 64 * 4;         // 98560

extern "C" void kernel_launch(void* const* d_in, const int* in_sizes, int n_in,
                              void* d_out, int out_size, void* d_ws, size_t ws_size,
                              hipStream_t stream) {
  (void)in_sizes; (void)n_in; (void)out_size; (void)ws_size;

  const float* cell_x = (const float*)d_in[0];
  const int*   eidx   = (const int*)d_in[1];
  const float* eattr  = (const float*)d_in[2];
  const float* mW1 = (const float*)d_in[3];
  const float* mb1 = (const float*)d_in[4];
  const float* mW2 = (const float*)d_in[5];
  const float* mb2 = (const float*)d_in[6];
  const float* uW1 = (const float*)d_in[7];
  const float* ub1 = (const float*)d_in[8];
  const float* uW2 = (const float*)d_in[9];
  const float* ub2 = (const float*)d_in[10];
  const float* gamma = (const float*)d_in[11];
  const float* beta  = (const float*)d_in[12];
  float* outp = (float*)d_out;

  char* ws = (char*)d_ws;
  _Float16* wT1 = (_Float16*)(ws + OFF_WT1);
  _Float16* wT2 = (_Float16*)(ws + OFF_WT2);
  _Float16* uT1 = (_Float16*)(ws + OFF_UT1);
  _Float16* uT2 = (_Float16*)(ws + OFF_UT2);
  float* aggp = (float*)(ws + OFF_AGG);
  float* cntp = (float*)(ws + OFF_CNT);

  (void)hipFuncSetAttribute(reinterpret_cast<const void*>(k_edge_msg),
                            hipFuncAttributeMaxDynamicSharedMemorySize, EDGE_SMEM);
  (void)hipFuncSetAttribute(reinterpret_cast<const void*>(k_node_update),
                            hipFuncAttributeMaxDynamicSharedMemorySize, NODE_SMEM);

  // weight prep (transpose + fp32 -> f16)
  k_transpose_cvt<<<(320 * 256 + 255) / 256, 256, 0, stream>>>(mW1, wT1, 320, 256);
  k_transpose_cvt<<<(256 * 256 + 255) / 256, 256, 0, stream>>>(mW2, wT2, 256, 256);
  k_transpose_cvt<<<(512 * 256 + 255) / 256, 256, 0, stream>>>(uW1, uT1, 512, 256);
  k_transpose_cvt<<<(256 * 256 + 255) / 256, 256, 0, stream>>>(uW2, uT2, 256, 256);

  // zero agg + cnt (contiguous in ws)
  const int zn = BSZ * N_CELLS * HID + BSZ * N_CELLS;
  k_zero_f32<<<(zn + 255) / 256, 256, 0, stream>>>(aggp, zn);

  // in-degree counts
  k_count<<<(BSZ * N_EDGES + 255) / 256, 256, 0, stream>>>(eidx, cntp);

  // edge message + scatter
  k_edge_msg<<<dim3(N_EDGES / 128, BSZ), 256, EDGE_SMEM, stream>>>(
      cell_x, eidx, eattr, wT1, mb1, wT2, mb2, aggp);

  // node update + residual layernorm
  k_node_update<<<dim3(N_CELLS / 64, BSZ), 128, NODE_SMEM, stream>>>(
      cell_x, aggp, cntp, uT1, ub1, uT2, ub2, gamma, beta, outp);
}
